// GAEModel_85676007620659
// MI455X (gfx1250) — compile-verified, matrix operand order checked by reading, not run on hardware
//
#include <hip/hip_runtime.h>
#include <hip/hip_bf16.h>

typedef __attribute__((ext_vector_type(16))) __bf16 v16bf;
typedef __attribute__((ext_vector_type(8)))  float  v8f;

static constexpr int NN   = 50000;   // nodes
static constexpr int NE   = 800000;  // edges
static constexpr int C_IN = 512;
static constexpr int C_H  = 256;
static constexpr int C_O  = 128;

__device__ __forceinline__ __bf16 f2bf(float f) {
  unsigned u = __builtin_bit_cast(unsigned, f);
  u = (u + 0x7FFFu + ((u >> 16) & 1u)) >> 16;          // round-to-nearest-even
  return __builtin_bit_cast(__bf16, (unsigned short)u);
}

// ---- degree / normalization -------------------------------------------------
__global__ void k_init_dis(float* dis) {
  int i = blockIdx.x * blockDim.x + threadIdx.x;
  if (i < NN) dis[i] = 1.0f;                            // self-loop contributes 1
}
__global__ void k_deg(const long long* __restrict__ dst, float* dis) {
  int e = blockIdx.x * blockDim.x + threadIdx.x;
  if (e < NE) atomicAdd(&dis[(int)dst[e]], 1.0f);
}
__global__ void k_finalize_dis(float* dis) {
  int i = blockIdx.x * blockDim.x + threadIdx.x;
  if (i < NN) dis[i] = rsqrtf(dis[i]);
}

// ---- fp32 -> bf16 shadow copy (optionally fused ReLU); 4 elems/thread -------
__global__ void k_cvt_bf16(const float* __restrict__ src, __bf16* __restrict__ dst,
                           size_t n4, int relu) {
  size_t t = (size_t)blockIdx.x * blockDim.x + threadIdx.x;
  if (t >= n4) return;
  float4 v = *(const float4*)(src + t * 4);
  if (relu) {
    v.x = fmaxf(v.x, 0.0f); v.y = fmaxf(v.y, 0.0f);
    v.z = fmaxf(v.z, 0.0f); v.w = fmaxf(v.w, 0.0f);
  }
  union { __bf16 b[4]; uint2 u; } p;
  p.b[0] = f2bf(v.x); p.b[1] = f2bf(v.y); p.b[2] = f2bf(v.z); p.b[3] = f2bf(v.w);
  *(uint2*)(dst + t * 4) = p.u;
}

// ---- WMMA GEMM: C[M,N] = A[M,K](bf16) * B[K,N](f32 weights)  ---------------
// Block = 8 waves; block tile = 128 rows x 32 cols. B strip staged bf16 in LDS
// (transposed, padded). Each wave: one 16-row tile x two 16-col tiles.
// Inner loop: 2x global b128 (A frag) + 4x ds b128 (B frags) + 2x WMMA.
// Requires K % 32 == 0, N % 32 == 0, M % 16 == 0.
__global__ __launch_bounds__(256) void k_wmma_gemm(
    const __bf16* __restrict__ A, const float* __restrict__ B,
    float* __restrict__ C, int M, int K, int N) {
  extern __shared__ __bf16 lds[];                // [32][K+8] = Bt[n][k]
  const int strideK = K + 8;                     // halves; rows stay 16B-aligned
  const int nBase   = blockIdx.y * 32;

  // cooperative stage: coalesced global reads (n fastest), bf16 convert once
  for (int idx = threadIdx.x; idx < 32 * K; idx += 256) {
    int n = idx & 31;
    int k = idx >> 5;
    lds[(size_t)n * strideK + k] = f2bf(B[(size_t)k * N + nBase + n]);
  }
  __syncthreads();

  const int wave = threadIdx.x >> 5, lane = threadIdx.x & 31;
  const int half = lane >> 4, m = lane & 15;     // ISA 7.12.2 16-bit layouts
  const int rowTileBase = blockIdx.x * 128 + wave * 16;
  if (rowTileBase >= M) return;                  // partial grid tail (post-barrier)

  const __bf16* arow = A + (size_t)(rowTileBase + m) * K;

  v8f acc[2] = {};
  for (int kb = 0; kb < K; kb += 32) {
    // A fragment: two contiguous 16B chunks (K-halves) -> 2x b128, no VALU glue
    const __bf16* ap = arow + kb + half * 8;
    union { v16bf v; uint4 q[2]; } au;
    au.q[0] = *(const uint4*)(ap);               // k = kb+half*8   .. +7
    au.q[1] = *(const uint4*)(ap + 16);          // k = kb+16+half*8 .. +7

#pragma unroll
    for (int t = 0; t < 2; ++t) {
      const __bf16* bp = lds + (size_t)(t * 16 + m) * strideK + kb + half * 8;
      union { v16bf v; uint4 q[2]; } bu;
      bu.q[0] = *(const uint4*)(bp);
      bu.q[1] = *(const uint4*)(bp + 16);
      acc[t] = __builtin_amdgcn_wmma_f32_16x16x32_bf16(false, au.v, false, bu.v,
                                                       (short)0, acc[t], false, false);
    }
  }

  // D layout: lanes 0-15 -> M=v, lanes 16-31 -> M=v+8; N = lane&15
  const int rbase = rowTileBase + half * 8;
#pragma unroll
  for (int t = 0; t < 2; ++t) {
    int n = nBase + t * 16 + m;
#pragma unroll
    for (int v = 0; v < 8; ++v)
      C[(size_t)(rbase + v) * N + n] = acc[t][v];
  }
}

// ---- self-loop message + bias: agg[i,c] = H[i,c]*dis[i]^2 + b[c] ------------
__global__ void k_self_bias(const float* __restrict__ H, const float* __restrict__ dis,
                            const float* __restrict__ b, float* __restrict__ agg, int C) {
  size_t t = (size_t)blockIdx.x * blockDim.x + threadIdx.x;
  if (t >= (size_t)NN * C) return;
  int i = (int)(t / C), c = (int)(t % C);
  float w = dis[i] * dis[i];
  agg[t] = H[t] * w + b[c];
}

// ---- edge scatter: agg[dst] += H[src] * dis[src]*dis[dst] -------------------
__global__ void k_scatter(const float* __restrict__ H, const float* __restrict__ dis,
                          const long long* __restrict__ src, const long long* __restrict__ dst,
                          float* __restrict__ agg, int C) {
  size_t t = (size_t)blockIdx.x * blockDim.x + threadIdx.x;
  int lanesPerEdge = C >> 2;                      // 4 channels / thread (float4)
  size_t e = t / lanesPerEdge;
  if (e >= NE) return;
  int c = (int)(t % lanesPerEdge) << 2;
  int s = (int)src[e], d = (int)dst[e];
  float w = dis[s] * dis[d];
  float4 hv = *(const float4*)(H + (size_t)s * C + c);
  float* o = agg + (size_t)d * C + c;
  atomicAdd(o + 0, hv.x * w);
  atomicAdd(o + 1, hv.y * w);
  atomicAdd(o + 2, hv.z * w);
  atomicAdd(o + 3, hv.w * w);
}

// ---- decoder: out[e] = sigmoid( Zw[src[e]] . z[dst[e]] + bb ) ---------------
__global__ void k_decode(const float* __restrict__ Zw, const float* __restrict__ z,
                         const long long* __restrict__ src, const long long* __restrict__ dst,
                         const float* __restrict__ bb, float* __restrict__ out) {
  size_t t = (size_t)blockIdx.x * blockDim.x + threadIdx.x;
  size_t e = t >> 5;
  int lane = (int)(t & 31);
  if (e >= NE) return;
  int s = (int)src[e], d = (int)dst[e];
  float4 a = *(const float4*)(Zw + (size_t)s * C_O + lane * 4);
  float4 b = *(const float4*)(z  + (size_t)d * C_O + lane * 4);
  float p = a.x * b.x + a.y * b.y + a.z * b.z + a.w * b.w;
#pragma unroll
  for (int off = 16; off > 0; off >>= 1) p += __shfl_xor(p, off, 32);
  if (lane == 0) out[e] = 1.0f / (1.0f + __expf(-(p + bb[0])));
}

extern "C" void kernel_launch(void* const* d_in, const int* in_sizes, int n_in,
                              void* d_out, int out_size, void* d_ws, size_t ws_size,
                              hipStream_t stream) {
  const float*     x   = (const float*)d_in[0];
  const long long* ei  = (const long long*)d_in[1];
  const float*     W1  = (const float*)d_in[2];
  const float*     b1  = (const float*)d_in[3];
  const float*     W2  = (const float*)d_in[4];
  const float*     b2  = (const float*)d_in[5];
  const float*     Wb  = (const float*)d_in[6];   // [1,128,128]
  const float*     bb  = (const float*)d_in[7];
  float*           out = (float*)d_out;

  const long long* src = ei;
  const long long* dst = ei + NE;

  // workspace layout (float units), with lifetime-based aliasing:
  //   regionA (NN*256 f): xbf (bf16, GEMM1 A)  -> later agg1 (f32)
  //   regionB (NN*256 f): H1 (f32)             -> later [hbf (bf16) | H2 (f32)]
  float* ws      = (float*)d_ws;
  float*  dis    = ws;                                      // 50000 (pad 50048)
  float*  regionA = ws + 50048;                             // NN*256 floats
  float*  regionB = regionA + (size_t)NN * 256;             // NN*256 floats
  float*  z      = regionB + (size_t)NN * 256;              // NN*128
  __bf16* zbf    = (__bf16*)(z + (size_t)NN * C_O);         // NN*128 bf16 (NN*64 f)
  float*  Zw     = (float*)(zbf + (size_t)NN * C_O);        // NN*128

  __bf16* xbf  = (__bf16*)regionA;                          // phase 1
  float*  agg1 = regionA;                                   // phase 2 (xbf dead)
  float*  H1   = regionB;                                   // phase 1-2
  __bf16* hbf  = (__bf16*)regionB;                          // phase 3 (H1 dead)
  float*  H2   = regionB + (size_t)NN * (C_H / 2);          // phase 3

  const int TB = 256;
  auto blocks = [](size_t n, int tb) { return (unsigned)((n + tb - 1) / tb); };
  auto gemm = [&](const __bf16* A, const float* B, float* C, int M, int K, int N) {
    dim3 grid((M + 127) / 128, N / 32);
    size_t shmem = (size_t)32 * (K + 8) * sizeof(__bf16);
    k_wmma_gemm<<<grid, TB, shmem, stream>>>(A, B, C, M, K, N);
  };

  // 1) symmetric normalization coefficients
  k_init_dis<<<blocks(NN, TB), TB, 0, stream>>>(dis);
  k_deg<<<blocks(NE, TB), TB, 0, stream>>>(dst, dis);
  k_finalize_dis<<<blocks(NN, TB), TB, 0, stream>>>(dis);

  // 2) layer 1: xbf = bf16(x); H1 = xbf @ W1 ; agg1 = scatter(norm*H1) + b1
  k_cvt_bf16<<<blocks((size_t)NN * C_IN / 4, TB), TB, 0, stream>>>(x, xbf,
                                                 (size_t)NN * C_IN / 4, 0);
  gemm(xbf, W1, H1, NN, C_IN, C_H);
  k_self_bias<<<blocks((size_t)NN * C_H, TB), TB, 0, stream>>>(H1, dis, b1, agg1, C_H);
  k_scatter<<<blocks((size_t)NE * (C_H / 4), TB), TB, 0, stream>>>(H1, dis, src, dst, agg1, C_H);

  // 3) layer 2: hbf = bf16(relu(agg1)); H2 = hbf @ W2 ; z = scatter(norm*H2) + b2
  k_cvt_bf16<<<blocks((size_t)NN * C_H / 4, TB), TB, 0, stream>>>(agg1, hbf,
                                                 (size_t)NN * C_H / 4, 1);
  gemm(hbf, W2, H2, NN, C_H, C_O);
  k_self_bias<<<blocks((size_t)NN * C_O, TB), TB, 0, stream>>>(H2, dis, b2, z, C_O);
  k_scatter<<<blocks((size_t)NE * (C_O / 4), TB), TB, 0, stream>>>(H2, dis, src, dst, z, C_O);

  // 4) hoisted bilinear: Zw = z @ Wb[0]  (since (z@Wb)[src] == z[src]@Wb)
  k_cvt_bf16<<<blocks((size_t)NN * C_O / 4, TB), TB, 0, stream>>>(z, zbf,
                                                 (size_t)NN * C_O / 4, 0);
  gemm(zbf, Wb, Zw, NN, C_O, C_O);

  // 5) per-edge dot + sigmoid
  k_decode<<<blocks((size_t)NE * 32, TB), TB, 0, stream>>>(Zw, z, src, dst, bb, out);
}
